// _DecoderLayer_62156766707863
// MI455X (gfx1250) — matrix operand order repaired and checked by reading, not compiled, not run on hardware
//
#include <hip/hip_runtime.h>
#include <hip/hip_bf16.h>
#include <math.h>

// ---------------- problem constants (match reference) ----------------
constexpr int Bc  = 16;
constexpr int LQc = 1024;
constexpr int LKc = 1024;
constexpr int DDc = 160;
constexpr int NHc = 8;
constexpr int HDc = 20;          // D / NH
constexpr int HPc = 32;          // padded head dim for WMMA K
constexpr int NTc = Bc * LQc;    // 16384 total tokens (== B*LK too)
constexpr float LOG2E = 1.44269504088896340736f;

typedef __bf16 bf16;
typedef __attribute__((ext_vector_type(16))) __bf16 v16bf;
typedef __attribute__((ext_vector_type(8)))  __bf16 v8bf;
typedef __attribute__((ext_vector_type(8)))  float  v8f;

__device__ __forceinline__ v8f wmma_bf16f32(v16bf a, v16bf b, v8f c) {
  // D = A(16x32 bf16) * B(32x16 bf16) + C(16x16 f32)
  return __builtin_amdgcn_wmma_f32_16x16x32_bf16(false, a, false, b,
                                                 (short)0, c, false, false);
}

// A-fragment: elements 0..15 = p[0..15] (16B-aligned p)
__device__ __forceinline__ v16bf ld_afrag(const bf16* p) {
  v8bf lo = *(const v8bf*)p;
  v8bf hi = *(const v8bf*)(p + 8);
  v16bf r;
#pragma unroll
  for (int e = 0; e < 8; ++e) { r[e] = lo[e]; r[8 + e] = hi[e]; }
  return r;
}
// B-fragment: 16 contiguous bf16 (16B-aligned p)
__device__ __forceinline__ v16bf ld_bfrag(const bf16* p) {
  v8bf lo = *(const v8bf*)p;
  v8bf hi = *(const v8bf*)(p + 8);
  v16bf r;
#pragma unroll
  for (int e = 0; e < 8; ++e) { r[e] = lo[e]; r[8 + e] = hi[e]; }
  return r;
}

// ---- DPP-based 16-lane butterfly reduction (no LDS, no waits) -------
// masks {xor1, xor2, mirror7, mirror15} generate the 16-lane group.
template <int CTRL>
__device__ __forceinline__ float dpp_movf(float x) {
  return __int_as_float(__builtin_amdgcn_update_dpp(
      0, __float_as_int(x), CTRL, 0xf, 0xf, true));
}
__device__ __forceinline__ float red16_max(float x) {
  x = fmaxf(x, dpp_movf<0xB1>(x));   // quad_perm [1,0,3,2]
  x = fmaxf(x, dpp_movf<0x4E>(x));   // quad_perm [2,3,0,1]
  x = fmaxf(x, dpp_movf<0x141>(x));  // row_half_mirror
  x = fmaxf(x, dpp_movf<0x140>(x));  // row_mirror
  return x;
}
__device__ __forceinline__ float red16_sum(float x) {
  x += dpp_movf<0xB1>(x);
  x += dpp_movf<0x4E>(x);
  x += dpp_movf<0x141>(x);
  x += dpp_movf<0x140>(x);
  return x;
}
// full 32-lane sum: 4 DPP stages + one cross-half shuffle
__device__ __forceinline__ float red32_sum(float x) {
  x = red16_sum(x);
  x += __shfl_xor(x, 16, 32);
  return x;
}

// ---------------- f32 -> bf16 conversion (weights) ----------------
__global__ void f32_to_bf16(const float* __restrict__ in,
                            bf16* __restrict__ out, int n) {
  int i = blockIdx.x * blockDim.x + threadIdx.x;
  if (i < n) out[i] = (bf16)in[i];
}

// ---------------- LayerNorm: f32 [rows, 160] -> bf16 ----------------
__global__ __launch_bounds__(32) void ln_to_bf16(
    const float* __restrict__ x, const float* __restrict__ g,
    const float* __restrict__ bta, bf16* __restrict__ y) {
  const int row  = blockIdx.x;
  const int lane = threadIdx.x;
  const float* xr = x + (size_t)row * DDc;
  float s = 0.f, s2 = 0.f;
  for (int i = lane; i < DDc; i += 32) { float v = xr[i]; s += v; s2 += v * v; }
  s  = red32_sum(s);
  s2 = red32_sum(s2);
  const float mean = s * (1.0f / DDc);
  const float var  = s2 * (1.0f / DDc) - mean * mean;
  const float inv  = rsqrtf(var + 1e-5f);
  bf16* yr = y + (size_t)row * DDc;
  for (int i = lane; i < DDc; i += 32)
    yr[i] = (bf16)((xr[i] - mean) * inv * g[i] + bta[i]);
}

// ---- head repack: [B*L, ld] (col slice, HD=20) -> [B,NH,L,32] padded.
// do_scale: fold softmax scale * log2(e) (incl. learnable temperature τ)
// into Q so the attention inner loop uses pure exp2 with no scaling.
__global__ void repack_qk(const bf16* __restrict__ in, int ld, int coloff,
                          bf16* __restrict__ out,
                          const float* __restrict__ log_tau, int do_scale) {
  int idx = blockIdx.x * blockDim.x + threadIdx.x;  // B*NH*1024*32
  int hd = idx & 31;
  int l  = (idx >> 5) & (LQc - 1);
  int h  = (idx >> 15) & (NHc - 1);
  int b  = idx >> 18;
  float sc = 1.0f;
  if (do_scale) {
    sc = rsqrtf((float)HDc) * LOG2E;  // base-2 softmax units
    if (log_tau) {  // tau = min(softplus(t)+0.5, 2.0)
      sc /= fminf(log1pf(__expf(log_tau[0])) + 0.5f, 2.0f);
    }
  }
  bf16 v = (bf16)0.0f;
  if (hd < HDc)
    v = (bf16)((float)in[(size_t)(b * LQc + l) * ld + coloff + h * HDc + hd] * sc);
  out[idx] = v;
}
// ---- V transposed repack: [B*LK, ld] -> [B,NH,32,LK] (pad rows zero)
__global__ void repack_vt(const bf16* __restrict__ in, int ld, int coloff,
                          bf16* __restrict__ out) {
  int idx = blockIdx.x * blockDim.x + threadIdx.x;  // B*NH*32*1024
  int k  = idx & (LKc - 1);
  int hd = (idx >> 10) & 31;
  int h  = (idx >> 15) & (NHc - 1);
  int b  = idx >> 18;
  bf16 v = (bf16)0.0f;
  if (hd < HDc) v = in[(size_t)(b * LKc + k) * ld + coloff + h * HDc + hd];
  out[idx] = v;
}

// ---------------- bf16 WMMA GEMM: out = A @ W^T + bias (+epilogue) ---
// A: [M, lda] bf16, W: [N, K] bf16 row-major. Wave tile 32(M) x 32(N).
// grid = (N/32, M/32). K compile-time -> fully scheduled K loop.
// EPI: 0=f32+bias, 1=bf16+bias, 2=f32+bias+resid, 3=bf16+bias+GELU.
template <int EPI, int K>
__global__ __launch_bounds__(32) void gemm_bf16_wmma(
    const bf16* __restrict__ A, int lda,
    const bf16* __restrict__ W,
    const float* __restrict__ bias,
    const float* __restrict__ resid,
    void* __restrict__ outv, int ldo) {
  const int lane = threadIdx.x & 31;
  const int half = lane >> 4;
  const int l16  = lane & 15;
  const int m0 = blockIdx.y << 5;
  const int n0 = blockIdx.x << 5;

  v8f acc[2][2] = {{{}, {}}, {{}, {}}};
#pragma unroll
  for (int k0 = 0; k0 < K; k0 += 32) {
    const bf16* ap = A + (size_t)(m0 + l16) * lda + k0 + 16 * half;
    v16bf a0 = ld_afrag(ap);
    v16bf a1 = ld_afrag(ap + (size_t)16 * lda);
    const bf16* wp = W + (size_t)(n0 + l16) * K + k0 + 16 * half;
    v16bf b0 = ld_bfrag(wp);
    v16bf b1 = ld_bfrag(wp + (size_t)16 * K);
    acc[0][0] = wmma_bf16f32(a0, b0, acc[0][0]);
    acc[0][1] = wmma_bf16f32(a0, b1, acc[0][1]);
    acc[1][0] = wmma_bf16f32(a1, b0, acc[1][0]);
    acc[1][1] = wmma_bf16f32(a1, b1, acc[1][1]);
  }
  // C layout: VGPR r, lanes 0-15 -> (M=r, N=lane); lanes 16-31 -> (M=r+8)
#pragma unroll
  for (int mi = 0; mi < 2; ++mi) {
#pragma unroll
    for (int r = 0; r < 8; ++r) {
      const int row = m0 + 16 * mi + r + 8 * half;
      const int c0 = n0 + l16;
      const int c1 = c0 + 16;
      float v0 = acc[mi][0][r] + bias[c0];
      float v1 = acc[mi][1][r] + bias[c1];
      if (EPI == 3) {  // exact GELU
        v0 = 0.5f * v0 * (1.0f + erff(v0 * 0.70710678118654752f));
        v1 = 0.5f * v1 * (1.0f + erff(v1 * 0.70710678118654752f));
      }
      if (EPI == 2) {
        v0 += resid[(size_t)row * ldo + c0];
        v1 += resid[(size_t)row * ldo + c1];
      }
      if (EPI == 0 || EPI == 2) {
        float* out = (float*)outv;
        out[(size_t)row * ldo + c0] = v0;
        out[(size_t)row * ldo + c1] = v1;
      } else {
        bf16* out = (bf16*)outv;
        out[(size_t)row * ldo + c0] = (bf16)v0;
        out[(size_t)row * ldo + c1] = (bf16)v1;
      }
    }
  }
}

// ---------------- Flash attention on padded head-major layouts -------
// Qp: [B,NH,LQ,32] (pre-scaled into log2 units), Kp: [B,NH,LK,32],
// Vt: [B,NH,32,LK]. One wave per (b, h, 32-row q tile); 32 keys/iter
// -> 8 WMMAs (4 S + 4 PV) per iteration sharing one set of K/V frags.
// grid = (LQ/32, NH, B), block = 32.
__global__ __launch_bounds__(32) void attn_wmma(
    const bf16* __restrict__ Qp, const bf16* __restrict__ Kp,
    const bf16* __restrict__ Vt, bf16* __restrict__ Ob, int ldo) {
  __shared__ alignas(16) bf16 Pl[32][32];
  const int lane = threadIdx.x & 31;
  const int half = lane >> 4;
  const int l16  = lane & 15;
  const int qt = blockIdx.x, h = blockIdx.y, b = blockIdx.z;

  const bf16* Q  = Qp + ((size_t)(b * NHc + h) * LQc + qt * 32) * HPc;
  const bf16* Kb = Kp + (size_t)(b * NHc + h) * LKc * HPc;
  const bf16* Vb = Vt + (size_t)(b * NHc + h) * HPc * LKc;
  bf16*       Op = Ob + (size_t)(b * LQc + qt * 32) * ldo + h * HDc;

  // Q A-fragments for the two 16-row tiles (rows 16B-aligned, padded)
  v16bf qa0 = ld_afrag(Q + (size_t)l16 * HPc + 16 * half);
  v16bf qa1 = ld_afrag(Q + (size_t)(16 + l16) * HPc + 16 * half);

  v8f o[2][2] = {{{}, {}}, {{}, {}}};
  float rm[2][8], rl[2][8];
#pragma unroll
  for (int mi = 0; mi < 2; ++mi)
#pragma unroll
    for (int r = 0; r < 8; ++r) { rm[mi][r] = -1e30f; rl[mi][r] = 0.0f; }

#pragma unroll 2
  for (int kt = 0; kt < LKc / 32; ++kt) {
    const int k0 = kt * 32;
    // ---- K fragments for keys k0..k0+15, k0+16..k0+31
    v16bf kb0 = ld_bfrag(Kb + (size_t)(k0 + l16) * HPc + 16 * half);
    v16bf kb1 = ld_bfrag(Kb + (size_t)(k0 + 16 + l16) * HPc + 16 * half);
    // ---- S tiles (Q pre-scaled; S already in log2 units)
    v8f s[2][2];
    s[0][0] = wmma_bf16f32(qa0, kb0, v8f{});
    s[0][1] = wmma_bf16f32(qa0, kb1, v8f{});
    s[1][0] = wmma_bf16f32(qa1, kb0, v8f{});
    s[1][1] = wmma_bf16f32(qa1, kb1, v8f{});

    // ---- online softmax per row (M = 16*mi + r + 8*half), base 2
#pragma unroll
    for (int mi = 0; mi < 2; ++mi) {
#pragma unroll
      for (int r = 0; r < 8; ++r) {
        float mx = red16_max(fmaxf(s[mi][0][r], s[mi][1][r]));
        float mnew = fmaxf(rm[mi][r], mx);
        float alpha = exp2f(rm[mi][r] - mnew);
        rm[mi][r] = mnew;
        float p0 = exp2f(s[mi][0][r] - mnew);
        float p1 = exp2f(s[mi][1][r] - mnew);
        s[mi][0][r] = p0;
        s[mi][1][r] = p1;
        float ps = red16_sum(p0 + p1);
        rl[mi][r] = rl[mi][r] * alpha + ps;
        o[mi][0][r] *= alpha;
        o[mi][1][r] *= alpha;
      }
    }

    // ---- P (32x32) through LDS: C-layout -> A-layout, full K=32
#pragma unroll
    for (int mi = 0; mi < 2; ++mi)
#pragma unroll
      for (int r = 0; r < 8; ++r) {
        Pl[16 * mi + r + 8 * half][l16]      = (bf16)s[mi][0][r];
        Pl[16 * mi + r + 8 * half][16 + l16] = (bf16)s[mi][1][r];
      }
    __syncthreads();
    v16bf pa0 = ld_afrag(&Pl[l16][0] + 16 * half);
    v16bf pa1 = ld_afrag(&Pl[16 + l16][0] + 16 * half);
    __syncthreads();

    // ---- O += P * V via transposed V: B[kk][n] = Vt[n][k0+kk]
    v16bf vb0 = ld_bfrag(Vb + (size_t)l16 * LKc + k0 + 16 * half);
    v16bf vb1 = ld_bfrag(Vb + (size_t)(16 + l16) * LKc + k0 + 16 * half);
    o[0][0] = wmma_bf16f32(pa0, vb0, o[0][0]);
    o[0][1] = wmma_bf16f32(pa0, vb1, o[0][1]);
    o[1][0] = wmma_bf16f32(pa1, vb0, o[1][0]);
    o[1][1] = wmma_bf16f32(pa1, vb1, o[1][1]);
  }

  // ---- normalize + store bf16 (head cols 0..19)
#pragma unroll
  for (int mi = 0; mi < 2; ++mi)
#pragma unroll
    for (int r = 0; r < 8; ++r) {
      float inv = 1.0f / rl[mi][r];
      int row = 16 * mi + r + 8 * half;
      Op[(size_t)row * ldo + l16] = (bf16)(o[mi][0][r] * inv);
      if (16 + l16 < HDc)
        Op[(size_t)row * ldo + 16 + l16] = (bf16)(o[mi][1][r] * inv);
    }
}

// =====================================================================
extern "C" void kernel_launch(void* const* d_in, const int* in_sizes, int n_in,
                              void* d_out, int out_size, void* d_ws,
                              size_t ws_size, hipStream_t stream) {
  (void)in_sizes; (void)n_in; (void)out_size; (void)ws_size;
  const float* q        = (const float*)d_in[0];
  const float* kv       = (const float*)d_in[1];
  const float* ln1_g    = (const float*)d_in[2];
  const float* ln1_b    = (const float*)d_in[3];
  const float* sa_in_w  = (const float*)d_in[4];
  const float* sa_in_b  = (const float*)d_in[5];
  const float* sa_out_w = (const float*)d_in[6];
  const float* sa_out_b = (const float*)d_in[7];
  const float* ln2_g    = (const float*)d_in[8];
  const float* ln2_b    = (const float*)d_in[9];
  const float* lnkv_g   = (const float*)d_in[10];
  const float* lnkv_b   = (const float*)d_in[11];
  const float* ca_qw    = (const float*)d_in[12];
  const float* ca_qb    = (const float*)d_in[13];
  const float* ca_kw    = (const float*)d_in[14];
  const float* ca_kb    = (const float*)d_in[15];
  const float* ca_vw    = (const float*)d_in[16];
  const float* ca_vb    = (const float*)d_in[17];
  const float* ca_ow    = (const float*)d_in[18];
  const float* ca_ob    = (const float*)d_in[19];
  const float* log_tau  = (const float*)d_in[20];
  const float* lnf_g    = (const float*)d_in[21];
  const float* lnf_b    = (const float*)d_in[22];
  const float* w1       = (const float*)d_in[23];
  const float* b1       = (const float*)d_in[24];
  const float* w2       = (const float*)d_in[25];
  const float* b2       = (const float*)d_in[26];

  // ---------------- workspace carving ----------------
  char* p = (char*)d_ws;
  auto carve = [&](size_t bytes) -> void* {
    void* r = (void*)p;
    p += (bytes + 255) & ~(size_t)255;
    return r;
  };
  const size_t headElems = (size_t)Bc * NHc * LQc * HPc;  // 4.19M per tensor
  bf16* wb_sa_in  = (bf16*)carve((size_t)3 * DDc * DDc * sizeof(bf16)); // 480x160
  bf16* wb_sa_out = (bf16*)carve((size_t)DDc * DDc * sizeof(bf16));
  bf16* wb_ca_q   = (bf16*)carve((size_t)DDc * DDc * sizeof(bf16));
  bf16* wb_ca_k   = (bf16*)carve((size_t)DDc * DDc * sizeof(bf16));
  bf16* wb_ca_v   = (bf16*)carve((size_t)DDc * DDc * sizeof(bf16));
  bf16* wb_ca_o   = (bf16*)carve((size_t)DDc * DDc * sizeof(bf16));
  bf16* wb_w1     = (bf16*)carve((size_t)4 * DDc * DDc * sizeof(bf16)); // 640x160
  bf16* wb_w2     = (bf16*)carve((size_t)DDc * 4 * DDc * sizeof(bf16)); // 160x640
  bf16* act0      = (bf16*)carve((size_t)NTc * DDc * sizeof(bf16));     // LN outputs
  bf16* qkv       = (bf16*)carve((size_t)NTc * 3 * DDc * sizeof(bf16)); // [NT,480]
  bf16* attn_o    = (bf16*)carve((size_t)NTc * DDc * sizeof(bf16));
  float* xres     = (float*)carve((size_t)NTc * DDc * sizeof(float));   // residual
  bf16* kvn       = (bf16*)carve((size_t)NTc * DDc * sizeof(bf16));
  // union region: padded Q/K/Vt (attention phases) vs FFN hidden buffer
  char* ureg = (char*)carve(3 * headElems * sizeof(bf16));  // 25.2MB >= 21MB
  bf16* Qp   = (bf16*)ureg;
  bf16* Kp   = Qp + headElems;
  bf16* Vtp  = Kp + headElems;
  bf16* hbuf = (bf16*)ureg;  // [NT,640], used only after attention is done
  // cq/ck/cv reuse the dead qkv region (exactly 3*160 columns)
  bf16* cq = qkv;
  bf16* ck = qkv + (size_t)NTc * DDc;
  bf16* cv = qkv + (size_t)2 * NTc * DDc;

  // ---------------- weight conversions f32 -> bf16 ----------------
  auto cvt = [&](const float* src, bf16* dst, int n) {
    f32_to_bf16<<<(n + 255) / 256, 256, 0, stream>>>(src, dst, n);
  };
  cvt(sa_in_w,  wb_sa_in,  3 * DDc * DDc);
  cvt(sa_out_w, wb_sa_out, DDc * DDc);
  cvt(ca_qw,    wb_ca_q,   DDc * DDc);
  cvt(ca_kw,    wb_ca_k,   DDc * DDc);
  cvt(ca_vw,    wb_ca_v,   DDc * DDc);
  cvt(ca_ow,    wb_ca_o,   DDc * DDc);
  cvt(w1,       wb_w1,     4 * DDc * DDc);
  cvt(w2,       wb_w2,     4 * DDc * DDc);

  const dim3 blk(32);
  const dim3 lnGrid(NTc);
  const dim3 attnGrid(LQc / 32, NHc, Bc);
  const int  rpBlocks = (int)(headElems / 256);

  // ---------------- self-attention ----------------
  ln_to_bf16<<<lnGrid, blk, 0, stream>>>(q, ln1_g, ln1_b, act0);
  gemm_bf16_wmma<1, DDc><<<dim3(3 * DDc / 32, NTc / 32), blk, 0, stream>>>(
      act0, DDc, wb_sa_in, sa_in_b, nullptr, qkv, 3 * DDc);
  // pad/repack heads out of fused qkv (ld=480); scale folded into Q
  repack_qk<<<rpBlocks, 256, 0, stream>>>(qkv, 3 * DDc, 0,   Qp, nullptr, 1);
  repack_qk<<<rpBlocks, 256, 0, stream>>>(qkv, 3 * DDc, DDc, Kp, nullptr, 0);
  repack_vt<<<rpBlocks, 256, 0, stream>>>(qkv, 3 * DDc, 2 * DDc, Vtp);
  attn_wmma<<<attnGrid, blk, 0, stream>>>(Qp, Kp, Vtp, attn_o, DDc);
  // x = q + attn_o @ sa_out_w^T + sa_out_b   (f32 residual stream)
  gemm_bf16_wmma<2, DDc><<<dim3(DDc / 32, NTc / 32), blk, 0, stream>>>(
      attn_o, DDc, wb_sa_out, sa_out_b, q, xres, DDc);

  // ---------------- cross-attention ----------------
  ln_to_bf16<<<lnGrid, blk, 0, stream>>>(xres, ln2_g, ln2_b, act0);
  ln_to_bf16<<<lnGrid, blk, 0, stream>>>(kv, lnkv_g, lnkv_b, kvn);
  gemm_bf16_wmma<1, DDc><<<dim3(DDc / 32, NTc / 32), blk, 0, stream>>>(
      act0, DDc, wb_ca_q, ca_qb, nullptr, cq, DDc);
  gemm_bf16_wmma<1, DDc><<<dim3(DDc / 32, NTc / 32), blk, 0, stream>>>(
      kvn, DDc, wb_ca_k, ca_kb, nullptr, ck, DDc);
  gemm_bf16_wmma<1, DDc><<<dim3(DDc / 32, NTc / 32), blk, 0, stream>>>(
      kvn, DDc, wb_ca_v, ca_vb, nullptr, cv, DDc);
  // scale (incl. learnable temperature) folded into Q at repack
  repack_qk<<<rpBlocks, 256, 0, stream>>>(cq, DDc, 0, Qp, log_tau, 1);
  repack_qk<<<rpBlocks, 256, 0, stream>>>(ck, DDc, 0, Kp, nullptr, 0);
  repack_vt<<<rpBlocks, 256, 0, stream>>>(cv, DDc, 0, Vtp);
  attn_wmma<<<attnGrid, blk, 0, stream>>>(Qp, Kp, Vtp, attn_o, DDc);
  // x += attn_o @ ca_ow^T + ca_ob  (in-place residual RMW per element)
  gemm_bf16_wmma<2, DDc><<<dim3(DDc / 32, NTc / 32), blk, 0, stream>>>(
      attn_o, DDc, wb_ca_o, ca_ob, xres, xres, DDc);

  // ---------------- FFN ----------------
  ln_to_bf16<<<lnGrid, blk, 0, stream>>>(xres, lnf_g, lnf_b, act0);
  gemm_bf16_wmma<3, DDc><<<dim3(4 * DDc / 32, NTc / 32), blk, 0, stream>>>(
      act0, DDc, wb_w1, b1, nullptr, hbuf, 4 * DDc);
  gemm_bf16_wmma<2, 4 * DDc><<<dim3(DDc / 32, NTc / 32), blk, 0, stream>>>(
      hbuf, 4 * DDc, wb_w2, b2, xres, d_out, DDc);
}
